// BNN_SW_38379827757533
// MI455X (gfx1250) — matrix operand order repaired and checked by reading, not compiled, not run on hardware
//
#include <hip/hip_runtime.h>
#include <hip/hip_bf16.h>
#include <stdint.h>

typedef __attribute__((ext_vector_type(16))) _Float16 v16h;
typedef __attribute__((ext_vector_type(8)))  _Float16 v8h;
typedef __attribute__((ext_vector_type(8)))  float    v8f;

// Problem constants
#define BATCH 16384
#define INF   784
#define KP    800     // INF padded to multiple of 32 (zero-padded)
#define HID   4096
#define OUTC  10
#define OUTP  16      // OUTC padded to 16 (zero weight rows)
#define NSTEP (KP / 32)

__device__ __forceinline__ _Float16 sgnh(float v) {
    return (v > 0.f) ? (_Float16)1.0f : ((v < 0.f) ? (_Float16)(-1.0f) : (_Float16)0.0f);
}

// Async global->LDS copy of 16 bytes (CDNA5, tracked with ASYNCcnt)
__device__ __forceinline__ void async_cp_b128(uint32_t lds_off, const void* gaddr) {
    asm volatile("global_load_async_to_lds_b128 %0, %1, off"
                 :: "v"(lds_off), "v"(gaddr) : "memory");
}
#define WAIT_ASYNC(n) asm volatile("s_wait_asynccnt %0" :: "i"(n) : "memory")

// ---- conversion kernels: build f16, K-padded staging buffers -------------

__global__ __launch_bounds__(256) void cvt_x(const float* __restrict__ x,
                                             _Float16* __restrict__ xh) {
    unsigned i = blockIdx.x * 256u + threadIdx.x;       // over BATCH*KP
    unsigned b = i / KP, k = i % KP;
    xh[i] = (k < INF) ? (_Float16)x[(size_t)b * INF + k] : (_Float16)0.0f;
}

__global__ __launch_bounds__(256) void cvt_w1(const float* __restrict__ W1,
                                              _Float16* __restrict__ w1s) {
    unsigned i = blockIdx.x * 256u + threadIdx.x;       // over HID*KP
    unsigned h = i / KP, k = i % KP;
    w1s[i] = (k < INF) ? sgnh(W1[(size_t)h * INF + k]) : (_Float16)0.0f;
}

__global__ __launch_bounds__(256) void cvt_w2(const float* __restrict__ W2,
                                              _Float16* __restrict__ w2s) {
    unsigned i = blockIdx.x * 256u + threadIdx.x;       // over OUTP*HID
    unsigned o = i / HID, k = i % HID;
    w2s[i] = (o < OUTC) ? sgnh(W2[(size_t)o * HID + k]) : (_Float16)0.0f;
}

// ---- GEMM1: h[B][HID] = x_f16 @ sign(W1)^T  (wave32 WMMA f16) ------------
// Block: 256 threads = 8 waves, tile 128(M) x 128(N), K-step 32.
// Wave layout: 2 (M) x 4 (N); each wave computes 64x32 = 4x2 WMMA tiles.
// Double-buffered LDS filled with global_load_async_to_lds_b128 (ASYNCcnt).
// LDS row stride = 40 halves (80 B): 16-B aligned, bank-conflict free.

#define LSTR 40

__global__ __launch_bounds__(256)
void gemm1_wmma(const _Float16* __restrict__ xh,
                const _Float16* __restrict__ w1s,
                _Float16* __restrict__ hbuf) {
    __shared__ __align__(16) _Float16 As[2][128 * LSTR];
    __shared__ __align__(16) _Float16 Bs[2][128 * LSTR];

    const int nTile = blockIdx.x & 31;
    const int mTile = blockIdx.x >> 5;
    const int m0 = mTile * 128, n0 = nTile * 128;
    const int tid = threadIdx.x;
    const int w = tid >> 5, lane = tid & 31;
    const int ln = lane & 15, hi = lane >> 4;
    const int wm = (w & 1) * 64, wn = (w >> 1) * 32;

    // This thread's two (row, chunk) assignments for each 128x32 tile fill:
    // 512 16-B chunks per matrix, 4 chunks per row.
    const int c0 = tid,        r0 = c0 >> 2, q0 = c0 & 3;
    const int c1 = tid + 256,  r1 = c1 >> 2, q1 = c1 & 3;

    // Issue 4 async b128 copies (2 for A, 2 for B) for K-step `kk` into buffer p.
    auto issue = [&](int kk, int p) {
        async_cp_b128((uint32_t)(uintptr_t)&As[p][r0 * LSTR + q0 * 8],
                      &xh[(size_t)(m0 + r0) * KP + kk + q0 * 8]);
        async_cp_b128((uint32_t)(uintptr_t)&Bs[p][r0 * LSTR + q0 * 8],
                      &w1s[(size_t)(n0 + r0) * KP + kk + q0 * 8]);
        async_cp_b128((uint32_t)(uintptr_t)&As[p][r1 * LSTR + q1 * 8],
                      &xh[(size_t)(m0 + r1) * KP + kk + q1 * 8]);
        async_cp_b128((uint32_t)(uintptr_t)&Bs[p][r1 * LSTR + q1 * 8],
                      &w1s[(size_t)(n0 + r1) * KP + kk + q1 * 8]);
    };

    v8f acc[4][2] = {};

    issue(0, 0);
    for (int s = 0; s < NSTEP; ++s) {
        const int p = s & 1;
        if (s + 1 < NSTEP) {
            issue((s + 1) * 32, p ^ 1);
            WAIT_ASYNC(4);   // oldest 4 (stage s) have landed; <=4 (stage s+1) in flight
        } else {
            WAIT_ASYNC(0);
        }
        __syncthreads();

        // A fragments: lanes 0-15 hold K{0..7,16..23}; lanes 16-31 K{8..15,24..31}
        v16h a[4];
#pragma unroll
        for (int i = 0; i < 4; ++i) {
            const _Float16* ap = &As[p][(wm + i * 16 + ln) * LSTR + hi * 8];
            v8h lo  = *reinterpret_cast<const v8h*>(ap);
            v8h hi8 = *reinterpret_cast<const v8h*>(ap + 16);
            a[i] = __builtin_shufflevector(lo, hi8, 0,1,2,3,4,5,6,7,8,9,10,11,12,13,14,15);
        }
        // B fragments: lanes 0-15 col n, K{0..7,16..23}; lanes 16-31 col n, K{8..15,24..31}
        v16h b[2];
#pragma unroll
        for (int j = 0; j < 2; ++j) {
            const _Float16* bp = &Bs[p][(wn + j * 16 + ln) * LSTR + hi * 8];
            v8h lo  = *reinterpret_cast<const v8h*>(bp);
            v8h hi8 = *reinterpret_cast<const v8h*>(bp + 16);
            b[j] = __builtin_shufflevector(lo, hi8, 0,1,2,3,4,5,6,7,8,9,10,11,12,13,14,15);
        }
#pragma unroll
        for (int i = 0; i < 4; ++i)
#pragma unroll
            for (int j = 0; j < 2; ++j)
                acc[i][j] = __builtin_amdgcn_wmma_f32_16x16x32_f16(
                    false, a[i], false, b[j], (short)0, acc[i][j], false, false);
        __syncthreads();
    }

    // C layout: VGPR r holds M = 8*hi + r (per 16x16 tile), N = ln
#pragma unroll
    for (int i = 0; i < 4; ++i)
#pragma unroll
        for (int j = 0; j < 2; ++j)
#pragma unroll
            for (int r = 0; r < 8; ++r) {
                int row = m0 + wm + i * 16 + hi * 8 + r;
                int col = n0 + wn + j * 16 + ln;
                hbuf[(size_t)row * HID + col] = (_Float16)acc[i][j][r];
            }
}

// ---- BN statistics -> per-column (scale, shift) --------------------------
// Block handles 64 columns; 4 row-stripes of 4096 rows per column, LDS reduce.

__global__ __launch_bounds__(256)
void bn_stats(const _Float16* __restrict__ hbuf,
              const float* __restrict__ gamma,
              const float* __restrict__ beta,
              float* __restrict__ scale,
              float* __restrict__ shift) {
    __shared__ float redS[256];
    __shared__ float redQ[256];
    const int cl = threadIdx.x & 63;
    const int s  = threadIdx.x >> 6;
    const int c  = blockIdx.x * 64 + cl;
    float sum = 0.f, sumsq = 0.f;
    for (int r = s * 4096; r < (s + 1) * 4096; ++r) {
        float v = (float)hbuf[(size_t)r * HID + c];
        sum += v; sumsq += v * v;
    }
    redS[threadIdx.x] = sum;
    redQ[threadIdx.x] = sumsq;
    __syncthreads();
    if (s == 0) {
        sum   = redS[cl] + redS[cl + 64] + redS[cl + 128] + redS[cl + 192];
        sumsq = redQ[cl] + redQ[cl + 64] + redQ[cl + 128] + redQ[cl + 192];
        float mean = sum * (1.f / BATCH);
        float var  = sumsq * (1.f / BATCH) - mean * mean;
        float sc   = gamma[c] * rsqrtf(var + 1e-5f);
        scale[c] = sc;
        shift[c] = beta[c] - mean * sc;
    }
}

// ---- GEMM2: logits = sign(h*scale+shift) @ sign(W2)^T --------------------
// Block: 256 threads = 8 waves; each wave one 16(M)x16(N) tile, K=4096.

__global__ __launch_bounds__(256)
void gemm2_wmma(const _Float16* __restrict__ hbuf,
                const float* __restrict__ scale,
                const float* __restrict__ shift,
                const _Float16* __restrict__ w2s,
                float* __restrict__ logits) {
    __shared__ __align__(16) float s_scale[HID];
    __shared__ __align__(16) float s_shift[HID];
    const int tid = threadIdx.x;
    for (int i = tid; i < HID; i += 256) { s_scale[i] = scale[i]; s_shift[i] = shift[i]; }
    __syncthreads();

    const int w = tid >> 5, lane = tid & 31, ln = lane & 15, hi = lane >> 4;
    const int row = blockIdx.x * 128 + w * 16 + ln;

    v8f acc = {};
    for (int k = 0; k < HID; k += 32) {
        v16h a;
#pragma unroll
        for (int c = 0; c < 2; ++c) {
            const int k0 = k + hi * 8 + c * 16;
            v8h hv = *reinterpret_cast<const v8h*>(&hbuf[(size_t)row * HID + k0]);
            const float4* scp = reinterpret_cast<const float4*>(&s_scale[k0]);
            const float4* shp = reinterpret_cast<const float4*>(&s_shift[k0]);
            float4 sc0 = scp[0], sc1 = scp[1];
            float4 sh0 = shp[0], sh1 = shp[1];
            float scs[8] = {sc0.x, sc0.y, sc0.z, sc0.w, sc1.x, sc1.y, sc1.z, sc1.w};
            float shs[8] = {sh0.x, sh0.y, sh0.z, sh0.w, sh1.x, sh1.y, sh1.z, sh1.w};
#pragma unroll
            for (int e = 0; e < 8; ++e) {
                float v = (float)hv[e] * scs[e] + shs[e];
                a[c * 8 + e] = sgnh(v);
            }
        }
        const _Float16* bp = &w2s[(size_t)ln * HID + k + hi * 8];
        v8h blo = *reinterpret_cast<const v8h*>(bp);
        v8h bhi = *reinterpret_cast<const v8h*>(bp + 16);
        v16h b = __builtin_shufflevector(blo, bhi, 0,1,2,3,4,5,6,7,8,9,10,11,12,13,14,15);
        acc = __builtin_amdgcn_wmma_f32_16x16x32_f16(
            false, a, false, b, (short)0, acc, false, false);
    }
#pragma unroll
    for (int r = 0; r < 8; ++r)
        logits[(size_t)(blockIdx.x * 128 + w * 16 + hi * 8 + r) * OUTP + ln] = acc[r];
}

// ---- log_softmax over 10 classes per row ---------------------------------

__global__ __launch_bounds__(256)
void lsm_kernel(const float* __restrict__ logits, float* __restrict__ out) {
    const int row = blockIdx.x * 256 + threadIdx.x;      // 16384 rows
    const float* l = &logits[(size_t)row * OUTP];
    float m = l[0];
#pragma unroll
    for (int c = 1; c < OUTC; ++c) m = fmaxf(m, l[c]);
    float s = 0.f;
#pragma unroll
    for (int c = 0; c < OUTC; ++c) s += expf(l[c] - m);
    float lse = m + logf(s);
#pragma unroll
    for (int c = 0; c < OUTC; ++c) out[(size_t)row * OUTC + c] = l[c] - lse;
}

// ---- launch --------------------------------------------------------------

extern "C" void kernel_launch(void* const* d_in, const int* in_sizes, int n_in,
                              void* d_out, int out_size, void* d_ws, size_t ws_size,
                              hipStream_t stream) {
    const float* x     = (const float*)d_in[0];
    const float* W1    = (const float*)d_in[1];
    const float* gamma = (const float*)d_in[2];
    const float* beta  = (const float*)d_in[3];
    const float* W2    = (const float*)d_in[4];
    float* out = (float*)d_out;

    // Workspace layout (all 16-B aligned by construction)
    char* ws = (char*)d_ws;
    _Float16* xh    = (_Float16*)ws;                         // BATCH*KP
    _Float16* w1s   = xh   + (size_t)BATCH * KP;             // HID*KP
    _Float16* w2s   = w1s  + (size_t)HID * KP;               // OUTP*HID
    _Float16* hbuf  = w2s  + (size_t)OUTP * HID;             // BATCH*HID
    float*    scale = (float*)(hbuf + (size_t)BATCH * HID);  // HID
    float*    shift = scale + HID;                           // HID
    float*    logit = shift + HID;                           // BATCH*OUTP

    cvt_x <<<(BATCH * KP) / 256, 256, 0, stream>>>(x, xh);
    cvt_w1<<<(HID * KP) / 256,   256, 0, stream>>>(W1, w1s);
    cvt_w2<<<(OUTP * HID) / 256, 256, 0, stream>>>(W2, w2s);

    gemm1_wmma<<<(BATCH / 128) * (HID / 128), 256, 0, stream>>>(xh, w1s, hbuf);
    bn_stats  <<<HID / 64, 256, 0, stream>>>(hbuf, gamma, beta, scale, shift);
    gemm2_wmma<<<BATCH / 128, 256, 0, stream>>>(hbuf, scale, shift, w2s, logit);
    lsm_kernel<<<BATCH / 256, 256, 0, stream>>>(logit, out);
}
